// ASTMiniGNN_4423816315484
// MI455X (gfx1250) — matrix operand order repaired and checked by reading, not compiled, not run on hardware
//
#include <hip/hip_runtime.h>
#include <hip/hip_bf16.h>
#include <stdint.h>

// Problem constants (match reference)
#define BB   8
#define NN   4096
#define EE   8190
#define VV   256
#define HH   128
#define OUTC 128
#define WPR  (NN / 32)          // 128 bitmap words per adjacency row

typedef __attribute__((ext_vector_type(16))) _Float16 v16h;
typedef __attribute__((ext_vector_type(8)))  _Float16 v8h;
typedef __attribute__((ext_vector_type(8)))  float    v8f;

// ---------------------------------------------------------------------------
// f32 -> f16 conversion (weights / embedding table)
// ---------------------------------------------------------------------------
__global__ void f32_to_f16_kernel(const float* __restrict__ src,
                                  _Float16* __restrict__ dst, int n) {
    int i = blockIdx.x * blockDim.x + threadIdx.x;
    if (i < n) dst[i] = (_Float16)src[i];
}

// ---------------------------------------------------------------------------
// Edge scatter: set bit (src,dst) in per-graph adjacency bitmap.
// atomicOr is idempotent -> duplicate edges collapse (matches .set semantics).
// ---------------------------------------------------------------------------
__global__ void scatter_kernel(const int* __restrict__ edges,
                               uint32_t* __restrict__ bitmap) {
    int idx = blockIdx.x * blockDim.x + threadIdx.x;
    if (idx >= BB * EE) return;
    int b = idx / EE;
    int e = idx - b * EE;
    const int* ep = edges + ((size_t)b * EE + e) * 2;
    int src = ep[0];
    int dst = ep[1];
    atomicOr(&bitmap[((size_t)b * NN + src) * WPR + (dst >> 5)],
             1u << (dst & 31));
}

// ---------------------------------------------------------------------------
// Degree + D^-1/2: one wave32 per adjacency row. deg = popcount(row) + 1 (+I).
// ---------------------------------------------------------------------------
__global__ __launch_bounds__(32)
void deg_kernel(const uint32_t* __restrict__ bitmap, float* __restrict__ dis) {
    const int bn   = blockIdx.x;          // 0 .. B*N-1
    const int lane = threadIdx.x;         // 0..31
    const uint32_t* row = bitmap + (size_t)bn * WPR;
    int cnt = 0;
    #pragma unroll
    for (int w = 0; w < WPR / 32; ++w) cnt += __popc(row[lane + 32 * w]);
    #pragma unroll
    for (int off = 16; off > 0; off >>= 1) cnt += __shfl_down(cnt, off, 32);
    if (lane == 0) {
        float deg = (float)(cnt + 1);     // +1 from +I (deg >= 1 always)
        dis[bn] = rsqrtf(deg);
    }
}

// ---------------------------------------------------------------------------
// WMMA GEMM: Y[r, o] = sum_k Af16[r, k] * W[o, k]   (torch Linear, no bias)
//   GATHER=true : Af16 row r comes from embh[ids[r] * H]  (fused embedding)
//   GATHER=false: Af16 row r comes from Asrc[r * H]
//
// Each wave owns FOUR 16x16 output tiles (one half of the H=128 output
// width) sharing one A-fragment per K-step -> 4 independent WMMA accumulator
// chains, so the WMMA->WMMA RAW hazard (up to 5 slots for f16) is hidden by
// interleaving instead of v_nop padding, and A-fragment traffic drops 4x.
// 16 x v_wmma_f32_16x16x32_f16 per wave.
//
// Fragment layouts used here (wave32; A/B k order contiguous, not ISA table):
//   A 16x32 f16 : lanes 0-15  -> M=lane,    h0..15 = K{kb+0..15}
//                 lanes 16-31 -> M=lane-16, h0..15 = K{kb+16..31}
//   B 32x16 f16 : lanes 0-15  -> N=lane,    h0..15 = K{kb+0..15}
//                 lanes 16-31 -> N=lane-16, h0..15 = K{kb+16..31}
//   C/D 16x16 f32: vgpr r, lane l -> M = r + (l<16?0:8), Ncol = l&15
// ---------------------------------------------------------------------------
template <bool GATHER>
__global__ __launch_bounds__(256)
void gemm_wmma_kernel(const _Float16* __restrict__ Asrc,
                      const int* __restrict__ ids,
                      const _Float16* __restrict__ W,
                      float* __restrict__ Yout,
                      int numRowTiles) {
    const int lane = threadIdx.x & 31;
    const int wave = threadIdx.x >> 5;
    const int seg  = blockIdx.x * 8 + wave;      // (rowTile, colQuad)
    const int colQuad = seg & 1;                 // 2 quads of 4 col-tiles
    const int tileRow = seg >> 1;
    if (tileRow >= numRowTiles) return;          // wave-uniform (EXEC stays full)

    const int halfSel = lane >> 4;               // 0: lanes 0-15, 1: lanes 16-31
    const int l15     = lane & 15;
    const int row     = tileRow * 16 + l15;

    const _Float16* arow;
    if (GATHER) arow = Asrc + (size_t)ids[row] * HH;
    else        arow = Asrc + (size_t)row * HH;

    const int aoff    = halfSel * 16;
    const int colBase = colQuad * 64 + l15;      // + ct*16 per tile

    // Per-lane W row pointers for the 4 column tiles
    const _Float16* wrow0 = W + (size_t)(colBase +  0) * HH + halfSel * 16;
    const _Float16* wrow1 = W + (size_t)(colBase + 16) * HH + halfSel * 16;
    const _Float16* wrow2 = W + (size_t)(colBase + 32) * HH + halfSel * 16;
    const _Float16* wrow3 = W + (size_t)(colBase + 48) * HH + halfSel * 16;

    v8f acc0 = {}, acc1 = {}, acc2 = {}, acc3 = {};
    #pragma unroll
    for (int kb = 0; kb < HH; kb += 32) {
        // shared A fragment for this K-step
        v8h a0 = *(const v8h*)(arow + kb + aoff);
        v8h a1 = *(const v8h*)(arow + kb + 8 + aoff);
        v16h av;
        #pragma unroll
        for (int i = 0; i < 8; ++i) { av[i] = a0[i]; av[8 + i] = a1[i]; }

        // 4 independent B fragments / accumulator chains
        v8h b0lo = *(const v8h*)(wrow0 + kb), b0hi = *(const v8h*)(wrow0 + kb + 8);
        v8h b1lo = *(const v8h*)(wrow1 + kb), b1hi = *(const v8h*)(wrow1 + kb + 8);
        v8h b2lo = *(const v8h*)(wrow2 + kb), b2hi = *(const v8h*)(wrow2 + kb + 8);
        v8h b3lo = *(const v8h*)(wrow3 + kb), b3hi = *(const v8h*)(wrow3 + kb + 8);
        v16h bv0, bv1, bv2, bv3;
        #pragma unroll
        for (int i = 0; i < 8; ++i) {
            bv0[i] = b0lo[i]; bv0[8 + i] = b0hi[i];
            bv1[i] = b1lo[i]; bv1[8 + i] = b1hi[i];
            bv2[i] = b2lo[i]; bv2[8 + i] = b2hi[i];
            bv3[i] = b3lo[i]; bv3[8 + i] = b3hi[i];
        }
        acc0 = __builtin_amdgcn_wmma_f32_16x16x32_f16(false, av, false, bv0,
                                                      (short)0, acc0, false, false);
        acc1 = __builtin_amdgcn_wmma_f32_16x16x32_f16(false, av, false, bv1,
                                                      (short)0, acc1, false, false);
        acc2 = __builtin_amdgcn_wmma_f32_16x16x32_f16(false, av, false, bv2,
                                                      (short)0, acc2, false, false);
        acc3 = __builtin_amdgcn_wmma_f32_16x16x32_f16(false, av, false, bv3,
                                                      (short)0, acc3, false, false);
    }

    float* yb = Yout + (size_t)(tileRow * 16 + halfSel * 8) * HH + colBase;
    #pragma unroll
    for (int r = 0; r < 8; ++r) {
        yb[(size_t)r * HH +  0] = acc0[r];
        yb[(size_t)r * HH + 16] = acc1[r];
        yb[(size_t)r * HH + 32] = acc2[r];
        yb[(size_t)r * HH + 48] = acc3[r];
    }
}

// ---------------------------------------------------------------------------
// Sparse aggregation: s[n, t] = dis[n] * ( sum_{bit(n,j)} dis[j]*Y[j,t]
//                                          + dis[n]*Y[n,t] )      (+I term)
// Layer 1: + b1, ReLU, store f16.  Layer 2: store f32.
// One 128-thread block per node row; bitmap row staged in LDS; the ctz
// bit-walk is uniform across the block (no divergence), Y reads coalesced.
// ---------------------------------------------------------------------------
__global__ __launch_bounds__(128)
void agg_kernel(const uint32_t* __restrict__ bitmap,
                const float* __restrict__ dis,
                const float* __restrict__ Yin,    // [B*N, H] f32
                const float* __restrict__ bias,   // b1 (layer1) or unused
                _Float16* __restrict__ Hout,      // layer1 out (f16) or null
                float* __restrict__ Zout) {       // layer2 out (f32) or null
    __shared__ uint32_t srow[WPR];
    const int t  = threadIdx.x;                   // channel 0..127
    const int bn = blockIdx.x;                    // 0..B*N-1
    const int b  = bn >> 12;                      // / N
    const int n  = bn & (NN - 1);

    srow[t] = bitmap[(size_t)bn * WPR + t];
    __syncthreads();

    const float disn = dis[bn];
    const float* Yb  = Yin + (size_t)b * NN * HH;
    const float* db  = dis + ((size_t)b << 12);

    float acc = 0.f;
    for (int w = 0; w < WPR; ++w) {
        uint32_t bits = srow[w];
        while (bits) {
            int j = (w << 5) + __builtin_ctz(bits);
            bits &= bits - 1;
            acc += db[j] * Yb[(size_t)j * HH + t];
        }
    }
    float v = disn * (acc + disn * Yb[(size_t)n * HH + t]);  // +I contribution
    if (Hout) {
        v += bias[t];
        v = v > 0.f ? v : 0.f;                                // ReLU
        Hout[(size_t)bn * HH + t] = (_Float16)v;
    } else {
        Zout[(size_t)bn * HH + t] = v;
    }
}

// ---------------------------------------------------------------------------
// Per-graph mean pool over N nodes, + b2, then L2-normalize (eps = 1e-12).
// One 512-thread block per graph; fully deterministic tree reductions.
// ---------------------------------------------------------------------------
__global__ __launch_bounds__(512)
void pool_norm_kernel(const float* __restrict__ Z,   // [B*N, OUT] f32
                      const float* __restrict__ b2,
                      float* __restrict__ out) {
    __shared__ float red[512];
    const int b = blockIdx.x;
    const int t = threadIdx.x;
    const int o = t & 127;
    const int s = t >> 7;                       // 4 slices over N
    const float* Zb = Z + (size_t)b * NN * HH;

    float acc = 0.f;
    for (int n = s; n < NN; n += 4) acc += Zb[(size_t)n * HH + o];
    red[t] = acc;
    __syncthreads();

    if (t < 128) {
        float v = red[t] + red[t + 128] + red[t + 256] + red[t + 384];
        v = v * (1.0f / NN) + b2[t];
        red[t]       = v;       // value
        red[128 + t] = v * v;   // square for norm
    }
    __syncthreads();
    for (int off = 64; off > 0; off >>= 1) {
        if (t < off) red[128 + t] += red[128 + t + off];
        __syncthreads();
    }
    if (t < 128) {
        float nrm = fmaxf(sqrtf(red[128]), 1e-12f);
        out[b * OUTC + t] = red[t] / nrm;
    }
}

// ---------------------------------------------------------------------------
// Host-side launch
// ---------------------------------------------------------------------------
extern "C" void kernel_launch(void* const* d_in, const int* in_sizes, int n_in,
                              void* d_out, int out_size, void* d_ws, size_t ws_size,
                              hipStream_t stream) {
    const int*   type_ids = (const int*)  d_in[0];   // [B, N]
    const int*   edges    = (const int*)  d_in[1];   // [B, E, 2]
    const float* emb      = (const float*)d_in[2];   // [V, H]
    const float* w1       = (const float*)d_in[3];   // [H, H]
    const float* b1       = (const float*)d_in[4];   // [H]
    const float* w2       = (const float*)d_in[5];   // [OUT, H]
    const float* b2       = (const float*)d_in[6];   // [OUT]
    float* out = (float*)d_out;                      // [B, OUT]

    // Workspace layout (256B-aligned offsets)
    char* ws = (char*)d_ws;
    const size_t OFF_BITMAP = 0;                                   // 16 MB
    const size_t SZ_BITMAP  = (size_t)BB * NN * WPR * sizeof(uint32_t);
    const size_t OFF_DIS    = OFF_BITMAP + SZ_BITMAP;              // 128 KB
    const size_t OFF_EMBH   = OFF_DIS  + (size_t)BB * NN * sizeof(float);
    const size_t OFF_W1H    = OFF_EMBH + (size_t)VV * HH * sizeof(_Float16);
    const size_t OFF_W2H    = OFF_W1H  + (size_t)HH * HH * sizeof(_Float16);
    const size_t OFF_Y      = OFF_W2H  + (size_t)HH * HH * sizeof(_Float16);
    const size_t OFF_HB     = OFF_Y    + (size_t)BB * NN * HH * sizeof(float);
    const size_t OFF_Z2     = OFF_HB   + (size_t)BB * NN * HH * sizeof(_Float16);

    uint32_t*  bitmap = (uint32_t*) (ws + OFF_BITMAP);
    float*     dis    = (float*)    (ws + OFF_DIS);
    _Float16*  embh   = (_Float16*) (ws + OFF_EMBH);
    _Float16*  w1h    = (_Float16*) (ws + OFF_W1H);
    _Float16*  w2h    = (_Float16*) (ws + OFF_W2H);
    float*     Y      = (float*)    (ws + OFF_Y);     // X@W1^T, later h@W2^T
    _Float16*  Hb     = (_Float16*) (ws + OFF_HB);    // relu layer-1 (f16)
    float*     Z2     = (float*)    (ws + OFF_Z2);    // layer-2 pre-pool

    // 1) clear adjacency bitmap
    hipMemsetAsync(bitmap, 0, SZ_BITMAP, stream);

    // 2) f32 -> f16 conversions
    f32_to_f16_kernel<<<(VV * HH + 255) / 256, 256, 0, stream>>>(emb, embh, VV * HH);
    f32_to_f16_kernel<<<(HH * HH + 255) / 256, 256, 0, stream>>>(w1, w1h, HH * HH);
    f32_to_f16_kernel<<<(HH * HH + 255) / 256, 256, 0, stream>>>(w2, w2h, HH * HH);

    // 3) edge scatter into bitmap
    scatter_kernel<<<(BB * EE + 255) / 256, 256, 0, stream>>>(edges, bitmap);

    // 4) degree -> D^-1/2
    deg_kernel<<<BB * NN, 32, 0, stream>>>(bitmap, dis);

    const int numRowTiles = BB * NN / 16;            // 2048
    const int gemmBlocks  = numRowTiles * 2 / 8;     // 512 (8 waves/blk, 4 tiles/wave)

    // 5) Y = emb[type_ids] @ w1^T   (WMMA, gather fused)
    gemm_wmma_kernel<true><<<gemmBlocks, 256, 0, stream>>>(
        embh, type_ids, w1h, Y, numRowTiles);

    // 6) Hb = relu(A @ Y + b1)      (sparse agg, f16 out)
    agg_kernel<<<BB * NN, 128, 0, stream>>>(bitmap, dis, Y, b1, Hb, nullptr);

    // 7) Y = Hb @ w2^T              (WMMA)
    gemm_wmma_kernel<false><<<gemmBlocks, 256, 0, stream>>>(
        Hb, nullptr, w2h, Y, numRowTiles);

    // 8) Z2 = A @ Y                 (sparse agg, f32 out)
    agg_kernel<<<BB * NN, 128, 0, stream>>>(bitmap, dis, Y, nullptr, nullptr, Z2);

    // 9) mean pool + b2 + L2 normalize
    pool_norm_kernel<<<BB, 512, 0, stream>>>(Z2, b2, out);
}
